// GraphMoE_80960133530409
// MI455X (gfx1250) — matrix operand order repaired and checked by reading, not compiled, not run on hardware
//
#include <hip/hip_runtime.h>
#include <stdint.h>

#define BB  4
#define SS  1024
#define DD  512
#define EE  8
#define KK2 2
#define DFF 2048
#define NT  (BB * SS)   // 4096 tokens

typedef __attribute__((ext_vector_type(16))) __bf16       v16bf;
typedef __attribute__((ext_vector_type(8)))  float        v8f;
typedef __attribute__((ext_vector_type(4)))  unsigned int u32x4;

union ABu { u32x4 u[2]; v16bf v; };

__device__ __forceinline__ unsigned short f2bf(float f) {
  unsigned u = __float_as_uint(f);
  unsigned r = u + 0x7FFFu + ((u >> 16) & 1u);   // round-to-nearest-even
  return (unsigned short)(r >> 16);
}

// branch-free tanh-approx GELU: tanh(z) = 1 - 2/(exp(2z)+1); result is rounded
// to bf16 right after, so hw v_exp_f32 + v_rcp_f32 precision is plenty.
__device__ __forceinline__ float gelu_tanh(float x) {
  const float c0 = 0.7978845608028654f;  // sqrt(2/pi)
  const float c1 = 0.044715f;
  float z  = c0 * (x + c1 * x * x * x);
  float t  = __expf(2.0f * z);
  float th = 1.0f - 2.0f * __builtin_amdgcn_rcpf(t + 1.0f);
  return 0.5f * x * (1.0f + th);
}

__device__ __forceinline__ float wave_sum(float v) {
#pragma unroll
  for (int off = 16; off > 0; off >>= 1) v += __shfl_xor(v, off, 32);
  return v;
}

// ---------------------------------------------------------------- utilities
__global__ void zero_kernel(float* p, int n) {
  int i = blockIdx.x * blockDim.x + threadIdx.x;
  if (i < n) p[i] = 0.0f;
}

__global__ void cvt_bf16(const float* __restrict__ in, unsigned short* __restrict__ out, int n) {
  for (int i = blockIdx.x * blockDim.x + threadIdx.x; i < n; i += gridDim.x * blockDim.x)
    out[i] = f2bf(in[i]);
}

// in: [E][R][C] f32  ->  out: [E][C][R] bf16
__global__ void transpose_cvt(const float* __restrict__ in, unsigned short* __restrict__ out,
                              int E, int R, int C) {
  size_t total = (size_t)E * R * C;
  for (size_t o = blockIdx.x * (size_t)blockDim.x + threadIdx.x; o < total;
       o += (size_t)gridDim.x * blockDim.x) {
    size_t e   = o / ((size_t)R * C);
    size_t rem = o - e * (size_t)R * C;
    size_t c   = rem / (size_t)R;
    size_t r   = rem - c * (size_t)R;
    out[o] = f2bf(in[(e * (size_t)R + r) * (size_t)C + c]);
  }
}

// ------------------------------------------------------------------- router
__global__ void router_local(const float* __restrict__ feat,
                             const float* __restrict__ w_local,
                             const float* __restrict__ b_local,
                             const float* __restrict__ w_gate,
                             const float* __restrict__ b_gate,
                             float* __restrict__ localS,
                             float* __restrict__ gateA) {
  int t = (blockIdx.x * blockDim.x + threadIdx.x) >> 5;
  int lane = threadIdx.x & 31;
  if (t >= NT) return;
  const float* x = feat + (size_t)t * DD;
  float acc[EE];
#pragma unroll
  for (int e = 0; e < EE; ++e) acc[e] = 0.0f;
  float ag = 0.0f;
  for (int d = lane; d < DD; d += 32) {
    float xv = x[d];
    const float* wl = w_local + (size_t)d * EE;
#pragma unroll
    for (int e = 0; e < EE; ++e) acc[e] += xv * wl[e];
    ag += xv * w_gate[d];
  }
#pragma unroll
  for (int e = 0; e < EE; ++e) acc[e] = wave_sum(acc[e]);
  ag = wave_sum(ag);
  if (lane == 0) {
#pragma unroll
    for (int e = 0; e < EE; ++e) localS[(size_t)t * EE + e] = acc[e] + b_local[e];
    gateA[t] = __builtin_amdgcn_rcpf(1.0f + __expf(-(ag + b_gate[0])));
  }
}

__global__ void router_combine(const float* __restrict__ adj,
                               const float* __restrict__ localS,
                               const float* __restrict__ gateA,
                               float* __restrict__ pweight,
                               int* __restrict__ topkOut,
                               float* __restrict__ usage) {
  int t = (blockIdx.x * blockDim.x + threadIdx.x) >> 5;
  int lane = threadIdx.x & 31;
  if (t >= NT) return;
  int b = t >> 10, i = t & 1023;
  const float* arow = adj + ((size_t)b * SS + i) * SS;
  float ns[EE];
#pragma unroll
  for (int e = 0; e < EE; ++e) ns[e] = 0.0f;
  float cnt = 0.0f;
  for (int j = lane; j < SS; j += 32) {
    float a = arow[j];
    if (a != 0.0f) {
      cnt += a;
      const float* ls = localS + ((size_t)b * SS + j) * EE;
#pragma unroll
      for (int e = 0; e < EE; ++e) ns[e] += a * ls[e];
    }
  }
#pragma unroll
  for (int e = 0; e < EE; ++e) ns[e] = wave_sum(ns[e]);
  cnt = wave_sum(cnt);
  if (lane == 0) {
    float g   = gateA[t];
    float inv = 1.0f / (cnt + 1e-8f);
    const float* ls = localS + (size_t)t * EE;
    float lg[EE], mx = -3.4e38f;
#pragma unroll
    for (int e = 0; e < EE; ++e) {
      lg[e] = g * ls[e] + (1.0f - g) * ns[e] * inv;
      mx = fmaxf(mx, lg[e]);
    }
    float p[EE], s = 0.0f;
#pragma unroll
    for (int e = 0; e < EE; ++e) { p[e] = __expf(lg[e] - mx); s += p[e]; }
    float rs = 1.0f / s;
#pragma unroll
    for (int e = 0; e < EE; ++e) p[e] *= rs;
    int i1 = 0;
#pragma unroll
    for (int e = 1; e < EE; ++e) if (p[e] > p[i1]) i1 = e;
    int i2 = (i1 == 0) ? 1 : 0;
#pragma unroll
    for (int e = 0; e < EE; ++e) if (e != i1 && p[e] > p[i2]) i2 = e;
    float den = p[i1] + p[i2] + 1e-8f;
    float* pw = pweight + (size_t)t * EE;
#pragma unroll
    for (int e = 0; e < EE; ++e) pw[e] = 0.0f;
    pw[i1] = p[i1] / den;
    pw[i2] = p[i2] / den;
    topkOut[t * KK2 + 0] = i1;
    topkOut[t * KK2 + 1] = i2;
#pragma unroll
    for (int e = 0; e < EE; ++e) atomicAdd(&usage[e], p[e]);
  }
}

__global__ void aux_kernel(const float* __restrict__ usage, float* __restrict__ outAux) {
  if (blockIdx.x == 0 && threadIdx.x == 0) {
    float u[EE], s = 0.0f;
#pragma unroll
    for (int e = 0; e < EE; ++e) { u[e] = usage[e]; s += u[e]; }
    float inv = 1.0f / (s + 1e-8f);
    float imp[EE], m = 0.0f;
#pragma unroll
    for (int e = 0; e < EE; ++e) { imp[e] = u[e] * inv; m += imp[e]; }
    m *= (1.0f / EE);
    float var = 0.0f;
#pragma unroll
    for (int e = 0; e < EE; ++e) { float d = imp[e] - m; var += d * d; }
    var *= (1.0f / EE);
    *outAux = sqrtf(var) / (m + 1e-8f);
  }
}

// ----------------------------------------------------- fused expert MLP (WMMA)
__global__ void __launch_bounds__(256)
moe_kernel(const unsigned short* __restrict__ featbf,
           const unsigned short* __restrict__ wupT,   // [E][DFF][DD] bf16
           const unsigned short* __restrict__ wdnT,   // [E][DD][DFF] bf16
           const float* __restrict__ b_up,            // [E][DFF]
           const float* __restrict__ b_down,          // [E][DD]
           const float* __restrict__ pweight,         // [NT][E]
           float* __restrict__ out) {
  extern __shared__ unsigned char smem[];
  unsigned short* xs   = (unsigned short*)smem;                    // 16KB
  unsigned short* hs   = (unsigned short*)(smem + 16384);          // 64KB
  float*          pwS  = (float*)(smem + 16384 + 65536);           // 512B
  int*            useE = (int*)(smem + 16384 + 65536 + 512);       // 32B

  const int t0   = blockIdx.x * 16;
  const int tid  = threadIdx.x;
  const int w    = tid >> 5;
  const int lane = tid & 31;
  const int col  = lane & 15;   // N column / A row (token)
  const int hh   = lane >> 4;   // lane-half selector for A/B K striping

  // ---- async stage x tile (16 x 512 bf16 = 16KB, 1024 x b128) into LDS ----
  {
    unsigned ldsBase = (unsigned)(uintptr_t)xs;
    unsigned long long gbase =
        (unsigned long long)(uintptr_t)(featbf + (size_t)t0 * DD);
#pragma unroll
    for (int c = 0; c < 4; ++c) {
      unsigned chunk = (unsigned)(tid * 4 + c);
      unsigned laddr = ldsBase + chunk * 16u;
      unsigned long long gaddr = gbase + (unsigned long long)chunk * 16ull;
      asm volatile("global_load_async_to_lds_b128 %0, %1, off"
                   :: "v"(laddr), "v"(gaddr) : "memory");
    }
  }
  if (tid < 128)
    pwS[tid] = pweight[(size_t)(t0 + (tid >> 3)) * EE + (tid & 7)];
  asm volatile("s_wait_asynccnt 0" ::: "memory");
  __syncthreads();
  if (tid < EE) {
    int any = 0;
#pragma unroll
    for (int r = 0; r < 16; ++r) any |= (pwS[r * EE + tid] != 0.0f);
    useE[tid] = any;
  }
  __syncthreads();

  float oacc[4][8];
#pragma unroll
  for (int a = 0; a < 4; ++a)
#pragma unroll
    for (int r = 0; r < 8; ++r) oacc[a][r] = 0.0f;

  for (int e = 0; e < EE; ++e) {
    if (!useE[e]) continue;   // block-uniform skip of unselected experts

    // ------------- up-projection: hs[16 x 2048] = gelu(x @ w_upT + b_up) ----
    {
      const unsigned short* wbase = wupT + (size_t)e * DFF * DD;
      for (int nt = 0; nt < 16; ++nt) {
        const int n = w * 256 + nt * 16 + col;
        const float bu = b_up[e * DFF + n];
        v8f acc = { bu, bu, bu, bu, bu, bu, bu, bu };
        const unsigned short* wrow = wbase + (size_t)n * DD;
        __builtin_prefetch(wrow + DD, 0, 1);
#pragma unroll 4
        for (int kk = 0; kk < 16; ++kk) {
          const int k0 = kk * 32;
          ABu a, b;
          const unsigned short* ap = xs + col * DD + k0 + hh * 8;
          a.u[0] = *(const u32x4*)(ap);
          a.u[1] = *(const u32x4*)(ap + 16);
          const unsigned short* bp = wrow + k0 + hh * 8;
          b.u[0] = *(const u32x4*)(bp);
          b.u[1] = *(const u32x4*)(bp + 16);
          acc = __builtin_amdgcn_wmma_f32_16x16x32_bf16(false, a.v, false, b.v,
                                                        (short)0, acc, false, false);
        }
#pragma unroll
        for (int r = 0; r < 8; ++r) {
          const int m = r + hh * 8;
          hs[m * DFF + n] = f2bf(gelu_tanh(acc[r]));
        }
      }
    }
    __syncthreads();

    // ------------- down-projection + weighted accumulate --------------------
    {
      const unsigned short* wbase = wdnT + (size_t)e * DD * DFF;
#pragma unroll
      for (int nt = 0; nt < 4; ++nt) {
        const int n = w * 64 + nt * 16 + col;
        const float bd = b_down[e * DD + n];
        v8f acc = { bd, bd, bd, bd, bd, bd, bd, bd };
        const unsigned short* wrow = wbase + (size_t)n * DFF;
        __builtin_prefetch(wrow + DFF, 0, 1);
#pragma unroll 4
        for (int kk = 0; kk < 64; ++kk) {
          const int k0 = kk * 32;
          ABu a, b;
          const unsigned short* ap = hs + col * DFF + k0 + hh * 8;
          a.u[0] = *(const u32x4*)(ap);
          a.u[1] = *(const u32x4*)(ap + 16);
          const unsigned short* bp = wrow + k0 + hh * 8;
          b.u[0] = *(const u32x4*)(bp);
          b.u[1] = *(const u32x4*)(bp + 16);
          acc = __builtin_amdgcn_wmma_f32_16x16x32_bf16(false, a.v, false, b.v,
                                                        (short)0, acc, false, false);
        }
#pragma unroll
        for (int r = 0; r < 8; ++r) {
          const int m = r + hh * 8;
          oacc[nt][r] += pwS[m * EE + e] * acc[r];
        }
      }
    }
    __syncthreads();
  }

#pragma unroll
  for (int nt = 0; nt < 4; ++nt) {
    const int n = w * 64 + nt * 16 + col;
#pragma unroll
    for (int r = 0; r < 8; ++r) {
      const int m = r + hh * 8;
      out[(size_t)(t0 + m) * DD + n] = oacc[nt][r];
    }
  }
}

// ---------------------------------------------------------------------------
extern "C" void kernel_launch(void* const* d_in, const int* in_sizes, int n_in,
                              void* d_out, int out_size, void* d_ws, size_t ws_size,
                              hipStream_t stream) {
  (void)in_sizes; (void)n_in; (void)out_size; (void)ws_size;
  const float* feat   = (const float*)d_in[0];
  const float* adj    = (const float*)d_in[1];
  const float* w_loc  = (const float*)d_in[2];
  const float* b_loc  = (const float*)d_in[3];
  const float* w_gate = (const float*)d_in[4];
  const float* b_gate = (const float*)d_in[5];
  const float* w_up   = (const float*)d_in[6];
  const float* b_up   = (const float*)d_in[7];
  const float* w_dn   = (const float*)d_in[8];
  const float* b_dn   = (const float*)d_in[9];

  char* ws = (char*)d_ws;
  size_t off = 0;
  auto carve = [&](size_t bytes) {
    char* p = ws + off;
    off = (off + bytes + 255) & ~(size_t)255;
    return p;
  };
  unsigned short* featbf = (unsigned short*)carve((size_t)NT * DD * 2);
  unsigned short* wupT   = (unsigned short*)carve((size_t)EE * DD * DFF * 2);
  unsigned short* wdnT   = (unsigned short*)carve((size_t)EE * DD * DFF * 2);
  float* localS  = (float*)carve((size_t)NT * EE * 4);
  float* gateA   = (float*)carve((size_t)NT * 4);
  float* pweight = (float*)carve((size_t)NT * EE * 4);
  float* usage   = (float*)carve(64);

  float* outMain = (float*)d_out;
  int*   outIdx  = (int*)d_out + (size_t)NT * DD;
  float* outAux  = (float*)d_out + (size_t)NT * DD + (size_t)NT * KK2;

  zero_kernel<<<1, 32, 0, stream>>>(usage, EE);
  cvt_bf16<<<1024, 256, 0, stream>>>(feat, featbf, NT * DD);
  transpose_cvt<<<4096, 256, 0, stream>>>(w_up, wupT, EE, DD, DFF);
  transpose_cvt<<<4096, 256, 0, stream>>>(w_dn, wdnT, EE, DFF, DD);
  router_local<<<NT / 8, 256, 0, stream>>>(feat, w_loc, b_loc, w_gate, b_gate,
                                           localS, gateA);
  router_combine<<<NT / 8, 256, 0, stream>>>(adj, localS, gateA, pweight,
                                             outIdx, usage);
  aux_kernel<<<1, 32, 0, stream>>>(usage, outAux);

  const size_t smem = 16384 + 65536 + 512 + 64;
  hipFuncSetAttribute(reinterpret_cast<const void*>(moe_kernel),
                      hipFuncAttributeMaxDynamicSharedMemorySize, (int)smem);
  moe_kernel<<<NT / 16, 256, 0, stream>>>(featbf, wupT, wdnT, b_up, b_dn,
                                          pweight, outMain);
}
